// YOLO_GNN_4698694221921
// MI455X (gfx1250) — matrix operand order repaired and checked by reading, not compiled, not run on hardware
//
#include <hip/hip_runtime.h>
#include <hip/hip_bf16.h>

// ---------------- problem constants ----------------
#define BATCH 16
#define S     196      // real patch count (14x14)
#define SP    224      // padded to multiple of 32 for WMMA K-loops
#define F     256
#define KPATCH 768     // 3*16*16
#define NC    80
#define HD    512
#define OD    256
#define TOPK  3

typedef __attribute__((ext_vector_type(16))) __bf16 v16bf;
typedef __attribute__((ext_vector_type(8)))  float  v8f;
typedef __attribute__((ext_vector_type(4)))  unsigned int u32x4;
typedef __attribute__((ext_vector_type(8)))  unsigned int u32x8;

// ---------------- WMMA helpers (gfx1250, wave32) ----------------
__device__ __forceinline__ v8f wmma_bf16(v16bf a, v16bf b, v8f c) {
  // D = A(16x32 bf16) x B(32x16 bf16) + C(16x16 f32)
  return __builtin_amdgcn_wmma_f32_16x16x32_bf16(false, a, false, b,
                                                 (short)0, c, false, false);
}

// A-matrix 16x32 bf16 fragment. `a` points at tile element (row0, k0),
// row-major with leading dim lda.
__device__ __forceinline__ v16bf lda_bf16(const __bf16* a, int lda, int lane) {
  int row  = lane & 15;
  int koff = (lane >> 4) << 4;            // 0 or 16
  const __bf16* p = a + (size_t)row * lda + koff;
  v16bf r;
#pragma unroll
  for (int v = 0; v < 8; ++v) {
    int k = 2 * v;
    r[2 * v]     = p[k];
    r[2 * v + 1] = p[k + 1];
  }
  return r;
}

// B-matrix 32x16 bf16 fragment, row-major source: B[k][n] = src[k*ldb + n].
__device__ __forceinline__ v16bf ldb_bf16_rm(const __bf16* src, int ldb, int lane) {
  int n  = lane & 15;
  int kb = (lane >> 4) << 4;              // 0 or 16
  const __bf16* p = src + (size_t)kb * ldb + n;
  v16bf r;
#pragma unroll
  for (int v = 0; v < 16; ++v) r[v] = p[(size_t)v * ldb];
  return r;
}

// B fragment where B[k][n] = src[n*ld + k] (Gram: B = featᵀ). K-contiguous.
__device__ __forceinline__ v16bf ldb_bf16_cm(const __bf16* src, int ldb, int lane) {
  int n  = lane & 15;
  int kb = (lane >> 4) << 4;
  const __bf16* p = src + (size_t)n * ldb + kb;
  v16bf r;
#pragma unroll
  for (int v = 0; v < 16; ++v) r[v] = p[v];
  return r;
}

// ---------------- TDM: 1-D async copy global -> LDS -------------------
// Builds a user-mode Tensor DMA Descriptor (D#) for a contiguous 1-D tile of
// `nelem` 2-byte elements and issues tensor_load_to_lds (2 SGPR groups: <=2D).
// Completion is tracked with TENSORcnt.
__device__ __forceinline__ void tdm_load_1d(unsigned lds_off, const void* gptr,
                                            unsigned nelem /* <= 65535 */) {
  unsigned long long ga = (unsigned long long)(size_t)gptr;
  u32x4 g0;
  g0[0] = 1u;                                   // count=1, user mode
  g0[1] = lds_off;                              // LDS byte address
  g0[2] = (unsigned)(ga & 0xFFFFFFFFu);         // global_addr[31:0]
  g0[3] = (unsigned)((ga >> 32) & 0x01FFFFFFu)  // global_addr[56:32]
        | (2u << 30);                           // type=2 ("image")
  u32x8 g1;
  g1[0] = (1u << 16);                           // wg_mask=0, data_size=1 (2B)
  g1[1] = (nelem & 0xFFFFu) << 16;              // tensor_dim0[15:0]
  g1[2] = (nelem >> 16) & 0xFFFFu;              // tensor_dim0[31:16], dim1=0
  g1[3] = (nelem & 0xFFFFu) << 16;              // tile_dim0 = nelem
  g1[4] = 0u;                                   // tile_dim1=0, tile_dim2=0
  g1[5] = nelem;                                // tensor_dim0_stride[31:0]
  g1[6] = 0u;
  g1[7] = 0u;
  asm volatile("tensor_load_to_lds %0, %1" :: "s"(g0), "s"(g1) : "memory");
}

// ---------------- K0: f32 -> bf16 weight pre-conversion ----------------
__global__ void k_cvt_bf16(const float* __restrict__ src,
                           __bf16* __restrict__ dst, int n) {
  int i = blockIdx.x * blockDim.x + threadIdx.x;
  if (i < n) dst[i] = (__bf16)src[i];
}

// ---------------- K1: patchify + feat = P @ patch_w + b ----------------
__global__ __launch_bounds__(256) void k_patch_gemm(
    const float* __restrict__ x, const __bf16* __restrict__ pwb,
    const float* __restrict__ pb, float* __restrict__ feat,
    __bf16* __restrict__ featb) {
  __shared__ __bf16 At[16 * KPATCH];      // 24 KB
  int b  = blockIdx.x / 14;
  int mt = blockIdx.x % 14;
  int tid = threadIdx.x;
  for (int idx = tid; idx < 16 * KPATCH; idx += 256) {
    int row = idx / KPATCH, k = idx % KPATCH;
    int p = mt * 16 + row;
    float v = 0.f;
    if (p < S) {
      int py = p / 14, px = p % 14;
      int c = k / 256, rem = k % 256, dy = rem / 16, dx = rem % 16;
      v = x[(((size_t)b * 3 + c) * 224 + (py * 16 + dy)) * 224 + (px * 16 + dx)];
    }
    At[idx] = (__bf16)v;
  }
  __syncthreads();
  int wave = tid >> 5, lane = tid & 31;
  v8f a0 = {}, a1 = {};
  int nt0 = wave, nt1 = wave + 8;
  for (int ks = 0; ks < KPATCH / 32; ++ks) {
    v16bf af = lda_bf16(At + ks * 32, KPATCH, lane);
    a0 = wmma_bf16(af, ldb_bf16_rm(pwb + (size_t)(ks * 32) * F + nt0 * 16, F, lane), a0);
    a1 = wmma_bf16(af, ldb_bf16_rm(pwb + (size_t)(ks * 32) * F + nt1 * 16, F, lane), a1);
  }
  int mb = (lane >> 4) << 3;
#pragma unroll
  for (int t = 0; t < 2; ++t) {
    int nt = t ? nt1 : nt0;
    int n = nt * 16 + (lane & 15);
    float bias = pb[n];
#pragma unroll
    for (int r = 0; r < 8; ++r) {
      int p = mt * 16 + mb + r;
      float v = (p < S) ? ((t ? a1[r] : a0[r]) + bias) : 0.f;  // zero pad rows
      size_t o = ((size_t)b * SP + p) * F + n;
      feat[o]  = v;
      featb[o] = (__bf16)v;
    }
  }
}

// ---------------- K2: mean-pool -> logits -> top-3 ----------------
__global__ __launch_bounds__(256) void k_cls_topk(
    const float* __restrict__ feat, const float* __restrict__ cw,
    const float* __restrict__ cb, int* __restrict__ topk) {
  __shared__ float mf[F];
  __shared__ float lg[NC];
  int b = blockIdx.x, f = threadIdx.x;
  float s = 0.f;
  for (int p = 0; p < S; ++p) s += feat[((size_t)b * SP + p) * F + f];
  mf[f] = s * (1.f / (float)S);
  __syncthreads();
  if (f < NC) {
    float a = cb[f];
    for (int k = 0; k < F; ++k) a += mf[k] * cw[k * NC + f];
    lg[f] = a;
  }
  __syncthreads();
  if (f == 0) {
    bool used[NC];
    for (int i = 0; i < NC; ++i) used[i] = false;
    for (int j = 0; j < TOPK; ++j) {       // repeated argmax == top_k tie rule
      int bi = 0; float bv = -3.4e38f;
      for (int c = 0; c < NC; ++c)
        if (!used[c] && lg[c] > bv) { bv = lg[c]; bi = c; }
      used[bi] = true;
      topk[b * TOPK + j] = bi;
    }
  }
}

// ---------------- K3: per-row squared norms ----------------
__global__ void k_sqnorm(const float* __restrict__ feat, float* __restrict__ sqn) {
  int row = blockIdx.x * blockDim.x + threadIdx.x;
  if (row >= BATCH * SP) return;
  const float* p = feat + (size_t)row * F;
  float s = 0.f;
  for (int f = 0; f < F; ++f) s += p[f] * p[f];
  sqn[row] = s;                             // pad rows are 0 -> 0
}

// ---------------- K4: Gram via WMMA -> d2 (padded entries = +inf) ----------
__global__ __launch_bounds__(256) void k_d2(
    const __bf16* __restrict__ featb, const float* __restrict__ sqn,
    float* __restrict__ d2) {
  int b = blockIdx.x / 14, it = blockIdx.x % 14;
  int wave = threadIdx.x >> 5, lane = threadIdx.x & 31;
  const __bf16* fb = featb + (size_t)b * SP * F;
  for (int jt = wave; jt < 14; jt += 8) {   // wave-uniform trip count
    v8f acc = {};
    for (int ks = 0; ks < F / 32; ++ks) {
      v16bf af = lda_bf16(fb + (size_t)(it * 16) * F + ks * 32, F, lane);
      v16bf bf = ldb_bf16_cm(fb + (size_t)(jt * 16) * F + ks * 32, F, lane);
      acc = wmma_bf16(af, bf, acc);
    }
    int n = lane & 15, mb = (lane >> 4) << 3;
#pragma unroll
    for (int r = 0; r < 8; ++r) {
      int i = it * 16 + mb + r, j = jt * 16 + n;
      float v;
      if (i < S && j < S) {
        v = sqn[b * SP + i] + sqn[b * SP + j] - 2.f * acc[r];
        v = v > 0.f ? v : 0.f;
      } else {
        v = 3.0e38f;                        // never selected by KNN
      }
      d2[((size_t)b * SP + i) * SP + j] = v;
    }
  }
}

// ---------------- K5: per-row 9-NN, drop self, exp weights, +I, normalize ---
__global__ __launch_bounds__(SP) void k_knn_adj(
    const float* __restrict__ d2, __bf16* __restrict__ Ab) {
  int b = blockIdx.x, s = threadIdx.x;
  __bf16* row = Ab + ((size_t)b * SP + s) * SP;
  if (s >= S) { for (int j = 0; j < SP; ++j) row[j] = (__bf16)0.f; return; }
  const float* dr = d2 + ((size_t)b * SP + s) * SP;
  float bv[9]; int bi[9];
#pragma unroll
  for (int i = 0; i < 9; ++i) { bv[i] = 3.4e38f; bi[i] = 0; }
  for (int j = 0; j < S; ++j) {             // 9 smallest, stable in index
    float v = dr[j];
    if (v < bv[8]) {
      int pos = 8;
      while (pos > 0 && v < bv[pos - 1]) {
        bv[pos] = bv[pos - 1]; bi[pos] = bi[pos - 1]; --pos;
      }
      bv[pos] = v; bi[pos] = j;
    }
  }
  float w[8]; float sum = 1.f;              // +1 from the identity term
#pragma unroll
  for (int i = 0; i < 8; ++i) { w[i] = __expf(-bv[i + 1]); sum += w[i]; }
  float inv = 1.f / sum;
  for (int j = 0; j < SP; ++j) row[j] = (__bf16)0.f;
#pragma unroll
  for (int i = 0; i < 8; ++i) row[bi[i + 1]] = (__bf16)(w[i] * inv);
  row[s] = (__bf16)inv;                     // self loop
}

// ---------------- K6: ax = A @ feat (WMMA, K=SP) ----------------
__global__ __launch_bounds__(256) void k_ax(
    const __bf16* __restrict__ Ab, const __bf16* __restrict__ featb,
    __bf16* __restrict__ axb) {
  int b = blockIdx.x / 14, mt = blockIdx.x % 14;
  int wave = threadIdx.x >> 5, lane = threadIdx.x & 31;
  const __bf16* Ar = Ab + (size_t)b * SP * SP + (size_t)(mt * 16) * SP;
  const __bf16* fb = featb + (size_t)b * SP * F;
  v8f a0 = {}, a1 = {};
  int nt0 = wave, nt1 = wave + 8;
  for (int ks = 0; ks < SP / 32; ++ks) {
    v16bf af = lda_bf16(Ar + ks * 32, SP, lane);
    a0 = wmma_bf16(af, ldb_bf16_rm(fb + (size_t)(ks * 32) * F + nt0 * 16, F, lane), a0);
    a1 = wmma_bf16(af, ldb_bf16_rm(fb + (size_t)(ks * 32) * F + nt1 * 16, F, lane), a1);
  }
  int mb = (lane >> 4) << 3;
#pragma unroll
  for (int t = 0; t < 2; ++t) {
    int n = (t ? nt1 : nt0) * 16 + (lane & 15);
#pragma unroll
    for (int r = 0; r < 8; ++r) {
      int p = mt * 16 + mb + r;
      axb[((size_t)b * SP + p) * F + n] = (__bf16)(t ? a1[r] : a0[r]);
    }
  }
}

// ---------------- K7: h = relu(ax @ W1[c] + b1[c]) per (b, topk) -----------
// W1 K-slabs (32x512 bf16 = 32 KB) are streamed into LDS by the Tensor Data
// Mover, double-buffered: slab ks+1 is in flight while WMMA consumes slab ks.
__global__ __launch_bounds__(256) void k_gnn1(
    const __bf16* __restrict__ axb, const __bf16* __restrict__ w1b,
    const float* __restrict__ b1, const int* __restrict__ topk,
    __bf16* __restrict__ hb) {
  __shared__ __bf16 Wt[2][32 * HD];         // 2 x 32 KB double buffer
  int blk = blockIdx.x;                     // BATCH*TOPK*14
  int mt = blk % 14; int bk = blk / 14; int kk = bk % TOPK; int b = bk / TOPK;
  int cls = topk[b * TOPK + kk];
  int wave = threadIdx.x >> 5, lane = threadIdx.x & 31;
  const __bf16* A0 = axb + (size_t)b * SP * F + (size_t)(mt * 16) * F;
  const __bf16* W = w1b + (size_t)cls * F * HD;
  const int KS = F / 32;                    // 8 k-steps
  if (wave == 0) {
    tdm_load_1d((unsigned)(size_t)(void*)&Wt[0][0], W, 32 * HD);
  }
  v8f acc[4] = {};
  for (int ks = 0; ks < KS; ++ks) {
    int cur = ks & 1;
    if (wave == 0) {
      if (ks + 1 < KS) {
        tdm_load_1d((unsigned)(size_t)(void*)&Wt[1 - cur][0],
                    W + (size_t)(ks + 1) * 32 * HD, 32 * HD);
        __builtin_amdgcn_s_wait_tensorcnt(1);   // slab `ks` done (in-order)
      } else {
        __builtin_amdgcn_s_wait_tensorcnt(0);
      }
    }
    __syncthreads();                        // publish slab `cur` to all waves
    v16bf af = lda_bf16(A0 + ks * 32, F, lane);
#pragma unroll
    for (int t = 0; t < 4; ++t) {
      int nt = wave * 4 + t;                // 32 N-tiles
      acc[t] = wmma_bf16(af, ldb_bf16_rm(&Wt[cur][nt * 16], HD, lane), acc[t]);
    }
    __syncthreads();                        // done reading before overwrite
  }
  int mb = (lane >> 4) << 3;
#pragma unroll
  for (int t = 0; t < 4; ++t) {
    int n = (wave * 4 + t) * 16 + (lane & 15);
    float bias = b1[cls * HD + n];
#pragma unroll
    for (int r = 0; r < 8; ++r) {
      int p = mt * 16 + mb + r;
      float v = acc[t][r] + bias;
      v = v > 0.f ? v : 0.f;
      if (p >= S) v = 0.f;                  // keep pad rows zero
      hb[((size_t)bk * SP + p) * HD + n] = (__bf16)v;
    }
  }
}

// ---------------- K8: ah = A @ h (WMMA, K=SP, N=HD) ----------------
__global__ __launch_bounds__(256) void k_ah(
    const __bf16* __restrict__ Ab, const __bf16* __restrict__ hb,
    __bf16* __restrict__ ahb) {
  int blk = blockIdx.x;                     // BATCH*TOPK*14
  int mt = blk % 14; int bk = blk / 14; int b = bk / TOPK;
  int wave = threadIdx.x >> 5, lane = threadIdx.x & 31;
  const __bf16* Ar = Ab + (size_t)b * SP * SP + (size_t)(mt * 16) * SP;
  const __bf16* H = hb + (size_t)bk * SP * HD;
  v8f acc[4] = {};
  for (int ks = 0; ks < SP / 32; ++ks) {
    v16bf af = lda_bf16(Ar + ks * 32, SP, lane);
#pragma unroll
    for (int t = 0; t < 4; ++t) {
      int nt = wave * 4 + t;
      acc[t] = wmma_bf16(af, ldb_bf16_rm(H + (size_t)(ks * 32) * HD + nt * 16, HD, lane), acc[t]);
    }
  }
  int mb = (lane >> 4) << 3;
#pragma unroll
  for (int t = 0; t < 4; ++t) {
    int n = (wave * 4 + t) * 16 + (lane & 15);
#pragma unroll
    for (int r = 0; r < 8; ++r) {
      int p = mt * 16 + mb + r;
      float v = (p < S) ? acc[t][r] : 0.f;
      ahb[((size_t)bk * SP + p) * HD + n] = (__bf16)v;
    }
  }
}

// ---------------- K9: out = ah @ W2[c] + b2[c], fused column-sum pooling ----
__global__ __launch_bounds__(256) void k_gnn2_pool(
    const __bf16* __restrict__ ahb, const __bf16* __restrict__ w2b,
    const float* __restrict__ b2, const int* __restrict__ topk,
    float* __restrict__ pooled) {
  int blk = blockIdx.x;                     // BATCH*TOPK*14
  int mt = blk % 14; int bk = blk / 14; int kk = bk % TOPK; int b = bk / TOPK;
  int cls = topk[b * TOPK + kk];
  int wave = threadIdx.x >> 5, lane = threadIdx.x & 31;
  const __bf16* A0 = ahb + (size_t)bk * SP * HD + (size_t)(mt * 16) * HD;
  const __bf16* W = w2b + (size_t)cls * HD * OD;
  v8f a0 = {}, a1 = {};
  int nt0 = wave, nt1 = wave + 8;           // 16 N-tiles
  for (int ks = 0; ks < HD / 32; ++ks) {
    v16bf af = lda_bf16(A0 + ks * 32, HD, lane);
    a0 = wmma_bf16(af, ldb_bf16_rm(W + (size_t)(ks * 32) * OD + nt0 * 16, OD, lane), a0);
    a1 = wmma_bf16(af, ldb_bf16_rm(W + (size_t)(ks * 32) * OD + nt1 * 16, OD, lane), a1);
  }
  int mb = (lane >> 4) << 3;
#pragma unroll
  for (int t = 0; t < 2; ++t) {
    int n = (t ? nt1 : nt0) * 16 + (lane & 15);
    float bias = b2[cls * OD + n];
    float colsum = 0.f;
#pragma unroll
    for (int r = 0; r < 8; ++r) {
      int p = mt * 16 + mb + r;
      if (p < S) colsum += (t ? a1[r] : a0[r]) + bias;
    }
    atomicAdd(&pooled[(size_t)bk * OD + n], colsum);
  }
}

// ---------------- K10: final = mean_k( pooled/196 @ fc_w + fc_b ) ----------
__global__ __launch_bounds__(128) void k_final(
    const float* __restrict__ pooled, const float* __restrict__ fcw,
    const float* __restrict__ fcb, float* __restrict__ out) {
  int b = blockIdx.x, c = threadIdx.x;
  if (c >= NC) return;
  float s = 0.f;
  for (int kk = 0; kk < TOPK; ++kk) {
    const float* p = pooled + ((size_t)b * TOPK + kk) * OD;
    float a = 0.f;
    for (int o = 0; o < OD; ++o) a += p[o] * fcw[o * NC + c];
    s += a;
  }
  out[b * NC + c] = fcb[c] + s * (1.f / (3.f * 196.f));
}

__global__ void k_zero(float* __restrict__ p, int n) {
  int i = blockIdx.x * blockDim.x + threadIdx.x;
  if (i < n) p[i] = 0.f;
}

// ---------------- launch ----------------
extern "C" void kernel_launch(void* const* d_in, const int* in_sizes, int n_in,
                              void* d_out, int out_size, void* d_ws, size_t ws_size,
                              hipStream_t stream) {
  (void)in_sizes; (void)n_in; (void)out_size; (void)ws_size;
  const float* x       = (const float*)d_in[0];
  const float* patch_w = (const float*)d_in[1];
  const float* patch_b = (const float*)d_in[2];
  const float* cls_w   = (const float*)d_in[3];
  const float* cls_b   = (const float*)d_in[4];
  const float* gnn_w1  = (const float*)d_in[5];
  const float* gnn_b1  = (const float*)d_in[6];
  const float* gnn_w2  = (const float*)d_in[7];
  const float* gnn_b2  = (const float*)d_in[8];
  const float* fc_w    = (const float*)d_in[9];
  const float* fc_b    = (const float*)d_in[10];
  float* out = (float*)d_out;

  char* wsp = (char*)d_ws;
  auto alloc = [&](size_t bytes) {
    char* p = wsp;
    wsp += (bytes + 255) & ~(size_t)255;
    return p;
  };
  float*  feat   = (float*) alloc((size_t)BATCH * SP * F * 4);
  __bf16* featb  = (__bf16*)alloc((size_t)BATCH * SP * F * 2);
  float*  sqn    = (float*) alloc((size_t)BATCH * SP * 4);
  float*  d2     = (float*) alloc((size_t)BATCH * SP * SP * 4);
  __bf16* Ab     = (__bf16*)alloc((size_t)BATCH * SP * SP * 2);
  __bf16* axb    = (__bf16*)alloc((size_t)BATCH * SP * F * 2);
  __bf16* hb     = (__bf16*)alloc((size_t)BATCH * TOPK * SP * HD * 2);
  __bf16* ahb    = (__bf16*)alloc((size_t)BATCH * TOPK * SP * HD * 2);
  float*  pooled = (float*) alloc((size_t)BATCH * TOPK * OD * 4);
  int*    topk   = (int*)   alloc((size_t)BATCH * TOPK * 4);
  __bf16* pwb    = (__bf16*)alloc((size_t)KPATCH * F * 2);
  __bf16* w1b    = (__bf16*)alloc((size_t)NC * F * HD * 2);
  __bf16* w2b    = (__bf16*)alloc((size_t)NC * HD * OD * 2);

  const int n_pw = KPATCH * F;
  const int n_w1 = NC * F * HD;
  const int n_w2 = NC * HD * OD;
  k_cvt_bf16  <<<(n_pw + 255) / 256, 256, 0, stream>>>(patch_w, pwb, n_pw);
  k_cvt_bf16  <<<(n_w1 + 255) / 256, 256, 0, stream>>>(gnn_w1, w1b, n_w1);
  k_cvt_bf16  <<<(n_w2 + 255) / 256, 256, 0, stream>>>(gnn_w2, w2b, n_w2);

  k_patch_gemm<<<BATCH * 14, 256, 0, stream>>>(x, pwb, patch_b, feat, featb);
  k_cls_topk  <<<BATCH, 256, 0, stream>>>(feat, cls_w, cls_b, topk);
  k_sqnorm    <<<(BATCH * SP + 255) / 256, 256, 0, stream>>>(feat, sqn);
  k_d2        <<<BATCH * 14, 256, 0, stream>>>(featb, sqn, d2);
  k_knn_adj   <<<BATCH, SP, 0, stream>>>(d2, Ab);
  k_ax        <<<BATCH * 14, 256, 0, stream>>>(Ab, featb, axb);
  k_gnn1      <<<BATCH * TOPK * 14, 256, 0, stream>>>(axb, w1b, gnn_b1, topk, hb);
  k_ah        <<<BATCH * TOPK * 14, 256, 0, stream>>>(Ab, hb, ahb);
  k_zero      <<<(BATCH * TOPK * OD + 255) / 256, 256, 0, stream>>>(pooled, BATCH * TOPK * OD);
  k_gnn2_pool <<<BATCH * TOPK * 14, 256, 0, stream>>>(ahb, w2b, gnn_b2, topk, pooled);
  k_final     <<<BATCH, 128, 0, stream>>>(pooled, fc_w, fc_b, out);
}
